// MultiHeadAttentionWithKVCache_32066225832237
// MI455X (gfx1250) — matrix operand order repaired and checked by reading, not compiled, not run on hardware
//
#include <hip/hip_runtime.h>
#include <hip/hip_bf16.h>

typedef __attribute__((ext_vector_type(16))) _Float16 v16h;
typedef __attribute__((ext_vector_type(8)))  _Float16 v8h;
typedef __attribute__((ext_vector_type(8)))  float    v8f;
typedef __attribute__((ext_vector_type(4)))  int      v4i;

constexpr int BB = 2;       // batch
constexpr int SS = 2048;    // sequence
constexpr int EE = 2048;    // embed
constexpr int HH = 16;      // heads
constexpr int DD = 128;     // head dim
constexpr int MM = BB * SS; // 4096 rows for projection GEMMs

#if __has_builtin(__builtin_amdgcn_global_load_async_to_lds_b128) && \
    __has_builtin(__builtin_amdgcn_s_wait_asynccnt)
#define USE_ASYNC_LDS 1
#else
#define USE_ASYNC_LDS 0
#endif

typedef __attribute__((address_space(1))) void* as1_ptr;
typedef __attribute__((address_space(3))) void* as3_ptr;
typedef __attribute__((address_space(1))) v4i*  as1_v4i;
typedef __attribute__((address_space(3))) v4i*  as3_v4i;

static __device__ __forceinline__ v16h cat16(v8h lo, v8h hi) {
  v16h r;
#pragma unroll
  for (int i = 0; i < 8; ++i) { r[i] = lo[i]; r[i + 8] = hi[i]; }
  return r;
}

template <int N> static __device__ __forceinline__ void wait_async() {
#if USE_ASYNC_LDS
  __builtin_amdgcn_s_wait_asynccnt(N);
#endif
}

// Copy one 16-byte chunk global->LDS (async engine if available).
static __device__ __forceinline__ void copy16(const _Float16* g, _Float16* l) {
#if USE_ASYNC_LDS
  __builtin_amdgcn_global_load_async_to_lds_b128(
      (as1_v4i)(as1_ptr)(void*)g, (as3_v4i)(as3_ptr)(void*)l, 0, 0);
#else
  *(v8h*)l = *(const v8h*)g;
#endif
}

template <typename T> struct alignas(4 * sizeof(T)) V4 { T x, y, z, w; };

// ---------------------------------------------------------------------------
// GEMM: out = A(M x E) * W(E x E) + bias, A is fp32 or f16, W/bias fp32.
// MODE 1: f16 out, (B,H,S,D) layout   (Q, K)
// MODE 2: f16 out, (B,H,D,S) layout   (V, pre-transposed for attention)
// MODE 3: f32 out, flat (B,S,E)       (final projection -> d_out)
// Block: 256 threads (8 waves), tile 128(M) x 64(N), BK=32.
// Each wave: 32x32 tile = 2x2 v_wmma_f32_16x16x32_f16 per K step.
// ---------------------------------------------------------------------------
template <typename TA, int MODE>
__global__ __launch_bounds__(256) void gemm_wmma(const TA* __restrict__ A,
                                                 const float* __restrict__ W,
                                                 const float* __restrict__ bias,
                                                 void* __restrict__ outp) {
  __shared__ _Float16 As[128][40];  // M x K tile (f16), +8 halves pad
  __shared__ _Float16 Bs[64][40];   // N x K tile (f16, transposed on fill)

  const int tid  = threadIdx.x;
  const int lane = tid & 31;
  const int wave = tid >> 5;
  const int wm   = wave & 3;   // wave row (4 in M)
  const int wn   = wave >> 2;  // wave col (2 in N)
  const int lrow = lane & 15;
  const int lsel = lane >> 4;
  const int rowBase = blockIdx.y * 128;
  const int colBase = blockIdx.x * 64;

  const int atr = tid >> 3;        // A fill: 32 rows / pass
  const int atc = (tid & 7) * 4;   // 4 elems along K
  const int wkr = tid >> 4;        // W fill: 16 k-rows / pass
  const int wnc = (tid & 15) * 4;  // 4 elems along N

  v8f acc[2][2] = {};

  for (int k0 = 0; k0 < EE; k0 += 32) {
    if (k0 + 32 < EE) {  // prefetch next tiles into cache (global_prefetch_b8)
      __builtin_prefetch(&A[(size_t)(rowBase + atr) * EE + k0 + 32 + atc], 0, 1);
      __builtin_prefetch(&W[(size_t)(k0 + 32 + wkr) * EE + colBase + wnc], 0, 1);
    }
    __syncthreads();
#pragma unroll
    for (int p = 0; p < 4; ++p) {  // A tile: 128x32, convert fp32/f16 -> f16
      const int r = p * 32 + atr;
      V4<TA> av = *(const V4<TA>*)&A[(size_t)(rowBase + r) * EE + k0 + atc];
      As[r][atc + 0] = (_Float16)av.x;
      As[r][atc + 1] = (_Float16)av.y;
      As[r][atc + 2] = (_Float16)av.z;
      As[r][atc + 3] = (_Float16)av.w;
    }
#pragma unroll
    for (int p = 0; p < 2; ++p) {  // W tile: 32x64, store transposed (N-major)
      const int kk = p * 16 + wkr;
      V4<float> wv = *(const V4<float>*)&W[(size_t)(k0 + kk) * EE + colBase + wnc];
      Bs[wnc + 0][kk] = (_Float16)wv.x;
      Bs[wnc + 1][kk] = (_Float16)wv.y;
      Bs[wnc + 2][kk] = (_Float16)wv.z;
      Bs[wnc + 3][kk] = (_Float16)wv.w;
    }
    __syncthreads();

    v16h af[2], bf[2];
#pragma unroll
    for (int i = 0; i < 2; ++i) {
      // A fragment (16x32 f16): lane holds row lrow; halves K = lsel*8.. / 16+lsel*8..
      const int r = wm * 32 + i * 16 + lrow;
      af[i] = cat16(*(const v8h*)&As[r][lsel * 8],
                    *(const v8h*)&As[r][16 + lsel * 8]);
      // B fragment (32x16 f16): lane holds col lrow; halves K = lsel*8.. / 16+lsel*8..
      const int c = wn * 32 + i * 16 + lrow;
      bf[i] = cat16(*(const v8h*)&Bs[c][lsel * 8],
                    *(const v8h*)&Bs[c][16 + lsel * 8]);
    }
#pragma unroll
    for (int i = 0; i < 2; ++i)
#pragma unroll
      for (int j = 0; j < 2; ++j)
        acc[i][j] = __builtin_amdgcn_wmma_f32_16x16x32_f16(
            false, af[i], false, bf[j], (short)0, acc[i][j], false, false);
  }

  // Epilogue: C layout -> row = v + 8*lsel, col = lrow
#pragma unroll
  for (int i = 0; i < 2; ++i) {
#pragma unroll
    for (int j = 0; j < 2; ++j) {
#pragma unroll
      for (int v = 0; v < 8; ++v) {
        const int gr = rowBase + wm * 32 + i * 16 + v + 8 * lsel;
        const int gc = colBase + wn * 32 + j * 16 + lrow;
        const float val = acc[i][j][v] + bias[gc];
        if (MODE == 3) {
          ((float*)outp)[(size_t)gr * EE + gc] = val;
        } else {
          const int bb = gr >> 11, s = gr & (SS - 1);
          const int h = gc >> 7, d = gc & (DD - 1);
          size_t idx;
          if (MODE == 1)       idx = (((size_t)bb * HH + h) * SS + s) * DD + d;
          else                 idx = (((size_t)bb * HH + h) * DD + d) * SS + s;
          ((_Float16*)outp)[idx] = (_Float16)val;
        }
      }
    }
  }
}

// ---------------------------------------------------------------------------
// Flash attention: per (b,h) head, streaming softmax over 32-key blocks.
// 128 threads = 4 waves; each wave owns 16 query rows and the full D=128
// f32 accumulator (8 C tiles). K/V tiles are double-buffered in LDS via
// async global->LDS copies (ASYNCcnt) so the copy of tile i+1 overlaps the
// 16 WMMAs of tile i.
// Q,K in (B,H,S,D) f16; V in (B,H,D,S) f16; out f16 flat (B,S,E).
// ---------------------------------------------------------------------------
constexpr int KROW = 136;  // K tile LDS row stride (128 + 8 pad halves)
constexpr int VROW = 40;   // V tile LDS row stride (32 + 8 pad halves)

// Stage one 32-key tile: K (32 x 128 f16) + V (128 x 32 f16) -> LDS.
// 1024 16-byte chunks split over 128 threads = 8 async ops per lane.
static __device__ __forceinline__ void stage_kv(const _Float16* __restrict__ Kg,
                                                const _Float16* __restrict__ Vg,
                                                _Float16* Ks, _Float16* Vs,
                                                int tid) {
#pragma unroll
  for (int p = 0; p < 4; ++p) {
    const int ck = tid + p * 128;            // K chunk: 32 rows x 16 chunks
    const int kr = ck >> 4, kc = (ck & 15) * 8;
    copy16(Kg + (size_t)kr * DD + kc, Ks + kr * KROW + kc);
    const int cv = tid + p * 128;            // V chunk: 128 rows x 4 chunks
    const int vr = cv >> 2, vc = (cv & 3) * 8;
    copy16(Vg + (size_t)vr * SS + vc, Vs + vr * VROW + vc);
  }
}

__global__ __launch_bounds__(128) void attn_wmma(const _Float16* __restrict__ Q,
                                                 const _Float16* __restrict__ K,
                                                 const _Float16* __restrict__ Vt,
                                                 _Float16* __restrict__ Aout) {
  __shared__ _Float16 Kt[2][32 * KROW];   // double-buffered K tile
  __shared__ _Float16 Vb[2][128 * VROW];  // double-buffered V tile
  __shared__ _Float16 Pl[4][16][40];      // per-wave P transpose patch

  const int tid  = threadIdx.x;
  const int lane = tid & 31;
  const int wave = tid >> 5;
  const int lrow = lane & 15;
  const int lsel = lane >> 4;
  const int bh   = blockIdx.y;
  const int b    = bh >> 4;
  const int h    = bh & 15;
  const int q0   = blockIdx.x * 64 + wave * 16;

  const _Float16* Qh = Q + (size_t)bh * SS * DD;
  const _Float16* Kh = K + (size_t)bh * SS * DD;
  const _Float16* Vh = Vt + (size_t)bh * DD * SS;

  // Q fragments for all 4 D-chunks, softmax scale 1/sqrt(128) folded in.
  const _Float16 qs = (_Float16)0.08838834764831845f;
  v16h qa[4];
#pragma unroll
  for (int c = 0; c < 4; ++c) {
    v8h lo = *(const v8h*)&Qh[(size_t)(q0 + lrow) * DD + c * 32 + lsel * 8];
    v8h hi = *(const v8h*)&Qh[(size_t)(q0 + lrow) * DD + c * 32 + 16 + lsel * 8];
#pragma unroll
    for (int i = 0; i < 8; ++i) { lo[i] *= qs; hi[i] *= qs; }
    qa[c] = cat16(lo, hi);
  }

  v8f o[8] = {};
  float m[8], l[8];
#pragma unroll
  for (int v = 0; v < 8; ++v) { m[v] = -3.0e38f; l[v] = 0.0f; }

  constexpr int NIT = SS / 32;
  stage_kv(Kh, Vh, Kt[0], Vb[0], tid);  // prologue: tile 0 in flight

  for (int it = 0; it < NIT; ++it) {
    const int cur = it & 1;
    if (it + 1 < NIT) {  // issue tile i+1 into the other buffer, then
      stage_kv(Kh + (size_t)(it + 1) * 32 * DD, Vh + (it + 1) * 32,
               Kt[cur ^ 1], Vb[cur ^ 1], tid);
      wait_async<8>();   // wait for tile i (oldest 8 ops), keep i+1 in flight
    } else {
      wait_async<0>();
    }
    __syncthreads();

    const _Float16* Ks = &Kt[cur][0];
    const _Float16* Vs = &Vb[cur][0];

    // --- scores: S = Q * K^T for two 16-key subtiles, accumulate over D ---
    v8f sc[2];
#pragma unroll
    for (int t = 0; t < 2; ++t) {
      v8f a = {};
#pragma unroll
      for (int c = 0; c < 4; ++c) {
        const _Float16* kp = Ks + (t * 16 + lrow) * KROW + c * 32 + lsel * 8;
        v16h bfr = cat16(*(const v8h*)kp, *(const v8h*)(kp + 16));
        a = __builtin_amdgcn_wmma_f32_16x16x32_f16(false, qa[c], false, bfr,
                                                   (short)0, a, false, false);
      }
      sc[t] = a;
    }

    // --- online softmax; row stats live per-VGPR (row = v + 8*lsel) ---
    float corr[8];
#pragma unroll
    for (int v = 0; v < 8; ++v) {
      float t0 = fmaxf(sc[0][v], sc[1][v]);
#pragma unroll
      for (int off = 1; off < 16; off <<= 1)
        t0 = fmaxf(t0, __shfl_xor(t0, off, 32));
      const float nm = fmaxf(m[v], t0);
      corr[v] = __expf(m[v] - nm);
      m[v] = nm;
      const float p0 = __expf(sc[0][v] - nm);
      const float p1 = __expf(sc[1][v] - nm);
      sc[0][v] = p0;
      sc[1][v] = p1;
      float rs = p0 + p1;
#pragma unroll
      for (int off = 1; off < 16; off <<= 1) rs += __shfl_xor(rs, off, 32);
      l[v] = l[v] * corr[v] + rs;
    }
#pragma unroll
    for (int t = 0; t < 8; ++t)
#pragma unroll
      for (int v = 0; v < 8; ++v) o[t][v] *= corr[v];

    // --- P: C layout -> A layout via per-wave LDS patch ---
#pragma unroll
    for (int v = 0; v < 8; ++v) {
      Pl[wave][v + 8 * lsel][lrow]      = (_Float16)sc[0][v];
      Pl[wave][v + 8 * lsel][16 + lrow] = (_Float16)sc[1][v];
    }
    asm volatile("s_wait_dscnt 0" ::: "memory");
    v16h pa = cat16(*(const v8h*)&Pl[wave][lrow][lsel * 8],
                    *(const v8h*)&Pl[wave][lrow][16 + lsel * 8]);

    // --- O += P * V : V tile is d-major so B-fragments are contiguous ---
#pragma unroll
    for (int t = 0; t < 8; ++t) {
      const _Float16* vp = Vs + (t * 16 + lrow) * VROW + lsel * 8;
      v16h bfr = cat16(*(const v8h*)vp, *(const v8h*)(vp + 16));
      o[t] = __builtin_amdgcn_wmma_f32_16x16x32_f16(false, pa, false, bfr,
                                                    (short)0, o[t], false, false);
    }
    __syncthreads();  // tile i fully consumed; its buffer may be refilled
  }

  // Normalize and scatter to flat (B,S,E) f16 for the output projection.
#pragma unroll
  for (int v = 0; v < 8; ++v) {
    const int srow = q0 + v + 8 * lsel;
    const float inv = 1.0f / l[v];
#pragma unroll
    for (int t = 0; t < 8; ++t) {
      Aout[((size_t)b * SS + srow) * EE + h * DD + t * 16 + lrow] =
          (_Float16)(o[t][v] * inv);
    }
  }
}

extern "C" void kernel_launch(void* const* d_in, const int* in_sizes, int n_in,
                              void* d_out, int out_size, void* d_ws,
                              size_t ws_size, hipStream_t stream) {
  (void)in_sizes; (void)n_in; (void)out_size; (void)ws_size;
  const float* q   = (const float*)d_in[0];
  const float* kin = (const float*)d_in[1];
  const float* vin = (const float*)d_in[2];
  const float* Wq  = (const float*)d_in[3];
  const float* bq  = (const float*)d_in[4];
  const float* Wk  = (const float*)d_in[5];
  const float* bk  = (const float*)d_in[6];
  const float* Wv  = (const float*)d_in[7];
  const float* bv  = (const float*)d_in[8];
  const float* Wo  = (const float*)d_in[9];
  const float* bo  = (const float*)d_in[10];

  const size_t elems = (size_t)BB * SS * EE;  // 8,388,608 halves = 16 MB each
  _Float16* Qws = (_Float16*)d_ws;            // (B,H,S,D)
  _Float16* Kws = Qws + elems;                // (B,H,S,D)
  _Float16* Vws = Kws + elems;                // (B,H,D,S)
  _Float16* Aws = Vws + elems;                // (B,S,E) attention output

  dim3 gg(EE / 64, MM / 128);  // 32 x 32 blocks
  gemm_wmma<float, 1><<<gg, 256, 0, stream>>>(q,   Wq, bq, Qws);
  gemm_wmma<float, 1><<<gg, 256, 0, stream>>>(kin, Wk, bk, Kws);
  gemm_wmma<float, 2><<<gg, 256, 0, stream>>>(vin, Wv, bv, Vws);
  attn_wmma<<<dim3(SS / 64, BB * HH), 128, 0, stream>>>(Qws, Kws, Vws, Aws);
  gemm_wmma<_Float16, 3><<<gg, 256, 0, stream>>>(Aws, Wo, bo, d_out);
}
